// GraphRNNSeq2SeqDecoder_64338610094363
// MI455X (gfx1250) — matrix operand order repaired and checked by reading, not compiled, not run on hardware
//
#include <hip/hip_runtime.h>
#include <stdint.h>

// ---------------------------------------------------------------------------
// Problem dimensions (compile-time, match reference)
// ---------------------------------------------------------------------------
constexpr int Bn = 16, Tn = 32, Sn = 512, Rn = 8, Ln = 256;
constexpr int Hn = 512, En = 512, Vn = 50000, NLn = 3, OOVn = 50;
constexpr int H3 = 3 * Hn;

// ---------------------------------------------------------------------------
// CDNA5 wave32 WMMA / TDM types
// ---------------------------------------------------------------------------
typedef __bf16 bf16_t;
typedef bf16_t v16bf __attribute__((ext_vector_type(16)));
typedef float  v8f   __attribute__((ext_vector_type(8)));
typedef unsigned int v4u __attribute__((ext_vector_type(4)));
typedef int v8i __attribute__((ext_vector_type(8)));
typedef int v4i __attribute__((ext_vector_type(4)));

__device__ __forceinline__ float sigm(float x) { return 1.0f / (1.0f + __expf(-x)); }

// ---------------------------------------------------------------------------
// TDM 2D tile load: global f32 tensor -> LDS (compacted tile), zero-fill OOB.
//   lds_addr     : byte offset of destination in LDS
//   gaddr        : VA of tile start inside the tensor
//   tile_w/h     : tile extent (elements / rows), tile_w*4 bytes per LDS row
//   tens_w/h     : remaining tensor extent from gaddr (for OOB zero-fill)
//   stride_elems : tensor row stride in elements
// Descriptor bitfields per CDNA5 ISA.
// This toolchain exposes the 6-arg builtin: (v4u, v8i, v4i, v4i, v8i, cpol).
// ---------------------------------------------------------------------------
__device__ __forceinline__ void tdm_load_2d(uint32_t lds_addr, uint64_t gaddr,
                                            uint32_t tile_w, uint32_t tile_h,
                                            uint32_t tens_w, uint32_t tens_h,
                                            uint32_t stride_elems) {
  v4u g0;
  g0[0] = 1u;                                            // count=1, user D#
  g0[1] = lds_addr;                                      // lds_addr [63:32]
  g0[2] = (uint32_t)(gaddr & 0xFFFFFFFFu);               // global_addr [95:64]
  g0[3] = (uint32_t)((gaddr >> 32) & 0x01FFFFFFu)        // global_addr [120:96]
        | (2u << 30);                                    // type = 2 (image)
  v8i g1;
  g1[0] = (int)(2u << 16);                               // data_size = 2 (4 bytes)
  g1[1] = (int)((tens_w & 0xFFFFu) << 16);               // tensor_dim0 [15:0]
  g1[2] = (int)((tens_w >> 16) | ((tens_h & 0xFFFFu) << 16)); // dim0 hi | dim1 lo
  g1[3] = (int)((tens_h >> 16) | (tile_w << 16));        // dim1 hi | tile_dim0
  g1[4] = (int)(tile_h & 0xFFFFu);                       // tile_dim1 (tile_dim2=0)
  g1[5] = (int)stride_elems;                             // tensor_dim0_stride lo
  g1[6] = 0;                                             // stride0 hi | stride1 lo
  g1[7] = 0;                                             // stride1 hi
  v4i z4 = {0, 0, 0, 0};                                 // 2D: groups 2/3 unused
  v8i z8 = {0, 0, 0, 0, 0, 0, 0, 0};
  __builtin_amdgcn_tensor_load_to_lds(g0, g1, z4, z4, z8, 0);
}

// ---------------------------------------------------------------------------
// Block-tiled, double-buffered, TDM-pipelined WMMA GEMM.
//   C[M,N] = A[M,K] * B + bias  (fp32 in/out, bf16 matrix core, f32 accum)
//   BT == 0 : B is [K,N] row-major        BT == 1 : B is [N,K] row-major (A@B^T)
// Workgroup: 256 threads = 8 waves as 4(M) x 2(N); each wave owns two adjacent
// 16x16 C tiles -> 64x64 macro-tile. TDM fills the next K-step's LDS buffers
// while the current one is consumed (single barrier per K-step).
// K must be a multiple of 32 (true for every call site).
// ---------------------------------------------------------------------------
template <int BT>
__global__ void wmma_gemm(const float* __restrict__ A, const float* __restrict__ Bm,
                          float* __restrict__ C, const float* __restrict__ bias,
                          int M, int N, int K, int lda, int ldb, int ldc) {
  __shared__ float As[2][64 * 32];       // A tile: 64 rows x 32 K   (2 x 8 KB)
  __shared__ float Bs[2][32 * 64];       // BT=0: [k][64] ; BT=1: [n][32] (2 x 8 KB)

  const int lane = threadIdx.x & 31;
  const int wave = threadIdx.x >> 5;     // 0..7
  const int wm = wave >> 1;              // M sub-tile 0..3
  const int wn = wave & 1;               // N half 0..1 (32 cols each)
  const int nt = (N + 63) >> 6;
  const int bm = blockIdx.x / nt;
  const int bnb = blockIdx.x % nt;
  const int m0 = bm << 6, n0 = bnb << 6;
  const int hf = lane >> 4;              // lane half
  const int lm = lane & 15;

  // Issue both TDM tile loads for K-step k0 into buffer `buf` (wave 0 only).
  auto issue = [&](int buf, int k0) {
    const uint64_t ga = (uint64_t)(uintptr_t)(A + (size_t)m0 * lda + k0);
    tdm_load_2d((uint32_t)(uintptr_t)(&As[buf][0]), ga, 32, 64,
                (uint32_t)(K - k0), (uint32_t)(M - m0), (uint32_t)lda);
    if (BT == 0) {                       // B[K,N]: rows = K, contiguous = N
      const uint64_t gb = (uint64_t)(uintptr_t)(Bm + (size_t)k0 * ldb + n0);
      tdm_load_2d((uint32_t)(uintptr_t)(&Bs[buf][0]), gb, 64, 32,
                  (uint32_t)(N - n0), (uint32_t)(K - k0), (uint32_t)ldb);
    } else {                             // B[N,K]: rows = N, contiguous = K
      const uint64_t gb = (uint64_t)(uintptr_t)(Bm + (size_t)n0 * ldb + k0);
      tdm_load_2d((uint32_t)(uintptr_t)(&Bs[buf][0]), gb, 32, 64,
                  (uint32_t)(K - k0), (uint32_t)(N - n0), (uint32_t)ldb);
    }
  };

  if (wave == 0) issue(0, 0);            // prime the pipeline

  v8f acc0 = {0.f, 0.f, 0.f, 0.f, 0.f, 0.f, 0.f, 0.f};
  v8f acc1 = {0.f, 0.f, 0.f, 0.f, 0.f, 0.f, 0.f, 0.f};

  for (int k0 = 0; k0 < K; k0 += 32) {
    const int cur = (k0 >> 5) & 1;
    if (wave == 0) __builtin_amdgcn_s_wait_tensorcnt(0);  // buf[cur] filled
    __syncthreads();                     // publish buf[cur]; buf[cur^1] now free
    if (wave == 0 && k0 + 32 < K) issue(cur ^ 1, k0 + 32);  // overlap with compute

    // A fragment 16x32 bf16: lanes 0-15 -> K {0..15}, 16-31 -> {16..31}
    v16bf av;
    const float* arow = &As[cur][(size_t)(wm * 16 + lm) * 32];
    const int ka = hf * 16;
#pragma unroll
    for (int v = 0; v < 8; ++v) {
      const int k = ka + (v << 1);
      const float2 f2 = *reinterpret_cast<const float2*>(arow + k);
      av[2 * v]     = (bf16_t)f2.x;
      av[2 * v + 1] = (bf16_t)f2.y;
    }
    // Two B fragments 32x16 bf16: cols wn*32 + {0..15} and wn*32 + {16..31}
    v16bf bv0, bv1;
    const int kb = hf << 4;              // VGPR j holds K = kb + {2j, 2j+1}
    if (BT == 0) {
      const float* b0 = &Bs[cur][wn * 32 + lm];
      const float* b1 = b0 + 16;
#pragma unroll
      for (int j = 0; j < 8; ++j) {
        bv0[2 * j]     = (bf16_t)b0[(size_t)(kb + 2 * j) * 64];
        bv0[2 * j + 1] = (bf16_t)b0[(size_t)(kb + 2 * j + 1) * 64];
        bv1[2 * j]     = (bf16_t)b1[(size_t)(kb + 2 * j) * 64];
        bv1[2 * j + 1] = (bf16_t)b1[(size_t)(kb + 2 * j + 1) * 64];
      }
    } else {
      const float* r0 = &Bs[cur][(size_t)(wn * 32 + lm) * 32];
      const float* r1 = &Bs[cur][(size_t)(wn * 32 + 16 + lm) * 32];
#pragma unroll
      for (int j = 0; j < 8; ++j) {
        const float2 f0 = *reinterpret_cast<const float2*>(r0 + kb + 2 * j);
        const float2 f1 = *reinterpret_cast<const float2*>(r1 + kb + 2 * j);
        bv0[2 * j]     = (bf16_t)f0.x;
        bv0[2 * j + 1] = (bf16_t)f0.y;
        bv1[2 * j]     = (bf16_t)f1.x;
        bv1[2 * j + 1] = (bf16_t)f1.y;
      }
    }
    // One A fragment feeds two independent WMMAs (back-to-back on the XDL pipe)
    acc0 = __builtin_amdgcn_wmma_f32_16x16x32_bf16(false, av, false, bv0,
                                                   (short)0, acc0, false, false);
    acc1 = __builtin_amdgcn_wmma_f32_16x16x32_bf16(false, av, false, bv1,
                                                   (short)0, acc1, false, false);
  }

  // C layout: VGPR j -> row (tile_m + hf*8 + j), col (tile_n + lane&15)
  const int cm0 = m0 + wm * 16 + hf * 8;
  const int cn0 = n0 + wn * 32 + lm;
  const int cn1 = cn0 + 16;
  if (cn0 < N) {
    const float badd = bias ? bias[cn0] : 0.f;
#pragma unroll
    for (int j = 0; j < 8; ++j) {
      const int cm = cm0 + j;
      if (cm < M) C[(size_t)cm * ldc + cn0] = acc0[j] + badd;
    }
  }
  if (cn1 < N) {
    const float badd = bias ? bias[cn1] : 0.f;
#pragma unroll
    for (int j = 0; j < 8; ++j) {
      const int cm = cm0 + j;
      if (cm < M) C[(size_t)cm * ldc + cn1] = acc1[j] + badd;
    }
  }
}

// ---------------------------------------------------------------------------
// Elementwise / reduction kernels
// ---------------------------------------------------------------------------
__global__ void k_copy(const float* __restrict__ src, float* __restrict__ dst, int n) {
  int i = blockIdx.x * blockDim.x + threadIdx.x;
  if (i < n) dst[i] = src[i];
}

__global__ void k_embed(const int* __restrict__ tokens, const float* __restrict__ embed_w,
                        float* __restrict__ x) {
  int i = blockIdx.x * blockDim.x + threadIdx.x;      // over B*T*E
  if (i >= Bn * Tn * En) return;
  int bt = i / En, e = i % En;
  x[i] = embed_w[(size_t)tokens[bt] * En + e];
}

// GRU gates: h = (1-z)*n + z*h_prev  (in-place update of layer hidden)
__global__ void k_gru_gate(const float* __restrict__ gx, const float* __restrict__ gh,
                           float* __restrict__ hlay) {
  int i = blockIdx.x * blockDim.x + threadIdx.x;      // over B*H
  if (i >= Bn * Hn) return;
  int b = i / Hn, h = i % Hn;
  const float* gxr = gx + (size_t)b * H3;
  const float* ghr = gh + (size_t)b * H3;
  float r = sigm(gxr[h] + ghr[h]);
  float z = sigm(gxr[Hn + h] + ghr[Hn + h]);
  float n = tanhf(gxr[2 * Hn + h] + r * ghr[2 * Hn + h]);
  hlay[i] = (1.f - z) * n + z * hlay[i];
}

// Fused source attention: scores (tanh dot v) -> masked softmax -> context.
// One block per batch element b. Writes raw attention weights into d_out attn region.
__global__ void k_src_attn(const float* __restrict__ enc_proj, const float* __restrict__ hp,
                           const float* __restrict__ vvec, const float* __restrict__ rep,
                           const unsigned char* __restrict__ mask,
                           float* __restrict__ ctx, float* __restrict__ aw_out, int t) {
  const int b = blockIdx.x, tid = threadIdx.x;
  __shared__ float shh[Hn];
  __shared__ float sc[Sn];
  __shared__ float red[256];
  for (int h = tid; h < Hn; h += 256) shh[h] = hp[b * Hn + h];
  __syncthreads();
  for (int s = tid; s < Sn; s += 256) {
    const float* ep = enc_proj + ((size_t)b * Sn + s) * Hn;
    float acc = 0.f;
    for (int h = 0; h < Hn; ++h) acc += tanhf(ep[h] + shh[h]) * vvec[h];
    sc[s] = mask[b * Sn + s] ? acc : -1e9f;
  }
  __syncthreads();
  float mx = -1e30f;
  for (int s = tid; s < Sn; s += 256) mx = fmaxf(mx, sc[s]);
  red[tid] = mx; __syncthreads();
  for (int st = 128; st > 0; st >>= 1) { if (tid < st) red[tid] = fmaxf(red[tid], red[tid + st]); __syncthreads(); }
  mx = red[0]; __syncthreads();
  float sum = 0.f;
  for (int s = tid; s < Sn; s += 256) sum += __expf(sc[s] - mx);
  red[tid] = sum; __syncthreads();
  for (int st = 128; st > 0; st >>= 1) { if (tid < st) red[tid] += red[tid + st]; __syncthreads(); }
  const float inv = 1.f / red[0]; __syncthreads();
  for (int s = tid; s < Sn; s += 256) {
    float a = __expf(sc[s] - mx) * inv;
    sc[s] = a;
    aw_out[((size_t)b * Tn + t) * Sn + s] = a;
  }
  __syncthreads();
  for (int h = tid; h < Hn; h += 256) {
    float acc = 0.f;
    const float* rp = rep + (size_t)b * Sn * Hn + h;
    for (int s = 0; s < Sn; ++s) acc += sc[s] * rp[(size_t)s * Hn];
    ctx[b * Hn + h] = acc;
  }
}

// Doc-level attention: one block, 128 threads (thread = (b,r)); softmax over R per b.
__global__ void k_doc_attn(const float* __restrict__ doc_proj, const float* __restrict__ hp,
                           const float* __restrict__ vvec, const unsigned char* __restrict__ mask,
                           float* __restrict__ ad) {
  const int tid = threadIdx.x;
  __shared__ float sc[Bn * Rn];
  if (tid < Bn * Rn) {
    const int b = tid / Rn;
    const float* dp = doc_proj + (size_t)tid * Hn;
    const float* hq = hp + b * Hn;
    float acc = 0.f;
    for (int h = 0; h < Hn; ++h) acc += tanhf(dp[h] + hq[h]) * vvec[h];
    sc[tid] = mask[tid] ? acc : -1e9f;
  }
  __syncthreads();
  if (tid < Bn) {
    float mx = -1e30f;
    for (int r = 0; r < Rn; ++r) mx = fmaxf(mx, sc[tid * Rn + r]);
    float sum = 0.f;
    for (int r = 0; r < Rn; ++r) sum += __expf(sc[tid * Rn + r] - mx);
    const float inv = 1.f / sum;
    for (int r = 0; r < Rn; ++r) ad[tid * Rn + r] = __expf(sc[tid * Rn + r] - mx) * inv;
  }
}

// Word-level attention scores (masked) : thread per (b,r,l)
__global__ void k_word_score(const float* __restrict__ word_proj, const float* __restrict__ hp,
                             const float* __restrict__ vvec, const unsigned char* __restrict__ mask,
                             float* __restrict__ scw) {
  int i = blockIdx.x * blockDim.x + threadIdx.x;      // over B*R*L
  if (i >= Bn * Rn * Ln) return;
  const int b = i / (Rn * Ln);
  const float* wp = word_proj + (size_t)i * Hn;
  const float* hq = hp + b * Hn;
  float acc = 0.f;
  for (int h = 0; h < Hn; ++h) acc += tanhf(wp[h] + hq[h]) * vvec[h];
  scw[i] = mask[i] ? acc : -1e9f;
}

// Softmax over L per (b,r), scaled by doc attention -> ra stored for step t
__global__ void k_word_softmax_ra(const float* __restrict__ scw, const float* __restrict__ ad,
                                  float* __restrict__ ra_all, int t) {
  const int br = blockIdx.x;                           // b*Rn + r
  const int b = br / Rn;
  const int tid = threadIdx.x;                         // 256 == Ln
  __shared__ float red[Ln];
  float v = scw[(size_t)br * Ln + tid];
  red[tid] = v; __syncthreads();
  for (int st = 128; st > 0; st >>= 1) { if (tid < st) red[tid] = fmaxf(red[tid], red[tid + st]); __syncthreads(); }
  const float mx = red[0]; __syncthreads();
  float e = __expf(v - mx);
  red[tid] = e; __syncthreads();
  for (int st = 128; st > 0; st >>= 1) { if (tid < st) red[tid] += red[tid + st]; __syncthreads(); }
  const float inv = 1.f / red[0];
  const int r = br % Rn;
  ra_all[(((size_t)b * Tn + t) * Rn + r) * Ln + tid] = ad[br] * e * inv;
}

// rctx[b,h] = sum_{r,l} ra[b,r,l] * ref_word_reps[b,r,l,h]
__global__ void k_rctx(const float* __restrict__ ra_all, const float* __restrict__ reps,
                       float* __restrict__ rctx, int t) {
  int i = blockIdx.x * blockDim.x + threadIdx.x;      // over B*H
  if (i >= Bn * Hn) return;
  const int b = i / Hn, h = i % Hn;
  const float* ra = ra_all + ((size_t)b * Tn + t) * Rn * Ln;
  const float* rp = reps + (size_t)b * Rn * Ln * Hn + h;
  float acc = 0.f;
  for (int rl = 0; rl < Rn * Ln; ++rl) acc += ra[rl] * rp[(size_t)rl * Hn];
  rctx[i] = acc;
}

// Fusion gate scalar per batch element
__global__ void k_gate_dot(const float* __restrict__ ctx, const float* __restrict__ rctx,
                           const float* __restrict__ gw, const float* __restrict__ gb,
                           float* __restrict__ gatev) {
  int b = threadIdx.x;
  if (b >= Bn) return;
  float s = gb[0];
  for (int h = 0; h < Hn; ++h) s += ctx[b * Hn + h] * gw[h];
  for (int h = 0; h < Hn; ++h) s += rctx[b * Hn + h] * gw[Hn + h];
  gatev[b] = sigm(s);
}

// context = g*ctx + (1-g)*rctx ; also record h_top for this step
__global__ void k_combine(const float* __restrict__ ctx, const float* __restrict__ rctx,
                          const float* __restrict__ gatev, const float* __restrict__ htop,
                          float* __restrict__ contexts, float* __restrict__ hiddens, int t) {
  int i = blockIdx.x * blockDim.x + threadIdx.x;      // over B*H
  if (i >= Bn * Hn) return;
  const int b = i / Hn, h = i % Hn;
  const float g = gatev[b];
  const size_t o = ((size_t)b * Tn + t) * Hn + h;
  contexts[o] = g * ctx[i] + (1.f - g) * rctx[i];
  hiddens[o]  = htop[i];
}

__global__ void k_build_ch(const float* __restrict__ contexts, const float* __restrict__ hiddens,
                           float* __restrict__ ch) {
  int i = blockIdx.x * blockDim.x + threadIdx.x;      // over B*T*H
  if (i >= Bn * Tn * Hn) return;
  const int row = i / Hn, h = i % Hn;
  ch[(size_t)row * (2 * Hn) + h]      = contexts[i];
  ch[(size_t)row * (2 * Hn) + Hn + h] = hiddens[i];
}

// pgen softmax over 3 mixture weights, thread per (b,t) row
__global__ void k_pgen(const float* __restrict__ contexts, const float* __restrict__ hiddens,
                       const float* __restrict__ x, const float* __restrict__ pw,
                       const float* __restrict__ pb, float* __restrict__ pg) {
  int row = blockIdx.x * blockDim.x + threadIdx.x;
  if (row >= Bn * Tn) return;
  float z0 = pb[0], z1 = pb[1], z2 = pb[2];
  const float* c = contexts + (size_t)row * Hn;
  const float* hh = hiddens + (size_t)row * Hn;
  const float* xx = x + (size_t)row * En;
  for (int i = 0; i < Hn; ++i) { const float* w = pw + (size_t)i * 3;           z0 += c[i] * w[0];  z1 += c[i] * w[1];  z2 += c[i] * w[2]; }
  for (int i = 0; i < Hn; ++i) { const float* w = pw + (size_t)(Hn + i) * 3;     z0 += hh[i] * w[0]; z1 += hh[i] * w[1]; z2 += hh[i] * w[2]; }
  for (int i = 0; i < En; ++i) { const float* w = pw + (size_t)(2 * Hn + i) * 3; z0 += xx[i] * w[0]; z1 += xx[i] * w[1]; z2 += xx[i] * w[2]; }
  float mx = fmaxf(z0, fmaxf(z1, z2));
  float e0 = __expf(z0 - mx), e1 = __expf(z1 - mx), e2 = __expf(z2 - mx);
  float inv = 1.f / (e0 + e1 + e2);
  pg[row * 3 + 0] = e0 * inv; pg[row * 3 + 1] = e1 * inv; pg[row * 3 + 2] = e2 * inv;
}

// In-place vocab softmax on d_out rows (logits already there, ldc = V+OOV),
// scaled by pg[...,0]; zero the OOV tail.
__global__ void k_vocab_softmax(float* __restrict__ outp, const float* __restrict__ pg) {
  const int row = blockIdx.x, tid = threadIdx.x;
  float* rowp = outp + (size_t)row * (Vn + OOVn);
  __shared__ float red[256];
  float mx = -1e30f;
  for (int i = tid; i < Vn; i += 256) mx = fmaxf(mx, rowp[i]);
  red[tid] = mx; __syncthreads();
  for (int st = 128; st > 0; st >>= 1) { if (tid < st) red[tid] = fmaxf(red[tid], red[tid + st]); __syncthreads(); }
  mx = red[0]; __syncthreads();
  float sum = 0.f;
  for (int i = tid; i < Vn; i += 256) sum += __expf(rowp[i] - mx);
  red[tid] = sum; __syncthreads();
  for (int st = 128; st > 0; st >>= 1) { if (tid < st) red[tid] += red[tid + st]; __syncthreads(); }
  const float scale = pg[row * 3] / red[0];
  for (int i = tid; i < Vn; i += 256) rowp[i] = __expf(rowp[i] - mx) * scale;
  for (int i = Vn + tid; i < Vn + OOVn; i += 256) rowp[i] = 0.f;
}

__global__ void k_scatter_src(float* __restrict__ outp, const float* __restrict__ aw,
                              const float* __restrict__ pg, const int* __restrict__ src_oov) {
  int i = blockIdx.x * blockDim.x + threadIdx.x;      // over B*T*S
  if (i >= Bn * Tn * Sn) return;
  const int b = i / (Tn * Sn), rem = i % (Tn * Sn);
  const int t = rem / Sn, s = rem % Sn;
  const int row = b * Tn + t;
  const float val = pg[row * 3 + 1] * aw[(size_t)row * Sn + s];
  atomicAdd(outp + (size_t)row * (Vn + OOVn) + src_oov[b * Sn + s], val);
}

__global__ void k_scatter_ref(float* __restrict__ outp, const float* __restrict__ ra_all,
                              const float* __restrict__ pg, const int* __restrict__ ref_oovs) {
  int i = blockIdx.x * blockDim.x + threadIdx.x;      // over B*T*R*L
  if (i >= Bn * Tn * Rn * Ln) return;
  const int b = i / (Tn * Rn * Ln), rem = i % (Tn * Rn * Ln);
  const int t = rem / (Rn * Ln), rl = rem % (Rn * Ln);
  const int row = b * Tn + t;
  const float val = pg[row * 3 + 2] * ra_all[(size_t)row * Rn * Ln + rl];
  atomicAdd(outp + (size_t)row * (Vn + OOVn) + ref_oovs[(size_t)b * Rn * Ln + rl], val);
}

// ---------------------------------------------------------------------------
// Host side
// ---------------------------------------------------------------------------
template <int BT>
static void gemm(hipStream_t st, const float* A, const float* Bm, float* C,
                 const float* bias, int M, int N, int K, int lda, int ldb, int ldc) {
  const int blocks = ((M + 63) / 64) * ((N + 63) / 64);   // 64x64 macro-tiles
  wmma_gemm<BT><<<blocks, 256, 0, st>>>(A, Bm, C, bias, M, N, K, lda, ldb, ldc);
}

extern "C" void kernel_launch(void* const* d_in, const int* in_sizes, int n_in,
                              void* d_out, int out_size, void* d_ws, size_t ws_size,
                              hipStream_t stream) {
  // ---- inputs (setup_inputs order) ----
  const int* tokens   = (const int*)d_in[0];
  const int* src_oov  = (const int*)d_in[1];
  const int* ref_oovs = (const int*)d_in[2];
  const unsigned char* src_mask      = (const unsigned char*)d_in[3];
  const unsigned char* ref_word_mask = (const unsigned char*)d_in[4];
  const unsigned char* ref_doc_mask  = (const unsigned char*)d_in[5];
  const float* cur_word_rep  = (const float*)d_in[7];
  const float* ref_word_reps = (const float*)d_in[8];
  const float* ref_doc_reps  = (const float*)d_in[9];
  const float* hidden0 = (const float*)d_in[10];
  const float* embed_w = (const float*)d_in[11];
  const float* gru_wih = (const float*)d_in[12];
  const float* gru_whh = (const float*)d_in[13];
  const float* gru_bih = (const float*)d_in[14];
  const float* gru_bhh = (const float*)d_in[15];
  const float* attn_Wd = (const float*)d_in[16];
  const float* attn_We = (const float*)d_in[17];
  const float* attn_v  = (const float*)d_in[18];
  const float* hd_Wd = (const float*)d_in[19];
  const float* hd_We = (const float*)d_in[20];
  const float* hd_v  = (const float*)d_in[21];
  const float* hw_Wd = (const float*)d_in[22];
  const float* hw_We = (const float*)d_in[23];
  const float* hw_v  = (const float*)d_in[24];
  const float* gate_w = (const float*)d_in[25];
  const float* gate_b = (const float*)d_in[26];
  const float* fc_w = (const float*)d_in[27];
  const float* fc_b = (const float*)d_in[28];
  const float* out_w = (const float*)d_in[29];
  const float* out_b = (const float*)d_in[30];
  const float* pgen_w = (const float*)d_in[31];
  const float* pgen_b = (const float*)d_in[32];

  float* out_final = (float*)d_out;                               // [B,T,V+OOV]
  float* out_attn  = out_final + (size_t)Bn * Tn * (Vn + OOVn);   // [B,T,S]

  // ---- workspace carve-up (floats) ----
  float* ws = (float*)d_ws;
  size_t off = 0;
  auto take = [&](size_t n) { float* p = ws + off; off += n; return p; };
  float* enc_proj  = take((size_t)Bn * Sn * Hn);
  float* doc_proj  = take((size_t)Bn * Rn * Hn);
  float* word_proj = take((size_t)Bn * Rn * Ln * Hn);
  float* xbuf      = take((size_t)Bn * Tn * En);
  float* hid       = take((size_t)NLn * Bn * Hn);
  float* gxb       = take((size_t)Bn * H3);
  float* ghb       = take((size_t)Bn * H3);
  float* hpa       = take((size_t)Bn * Hn);
  float* hpd       = take((size_t)Bn * Hn);
  float* hpw       = take((size_t)Bn * Hn);
  float* ctx       = take((size_t)Bn * Hn);
  float* rctx      = take((size_t)Bn * Hn);
  float* gatev     = take(64);
  float* scw       = take((size_t)Bn * Rn * Ln);
  float* adw       = take((size_t)Bn * Rn);
  float* ra_all    = take((size_t)Bn * Tn * Rn * Ln);
  float* contexts  = take((size_t)Bn * Tn * Hn);
  float* hiddens   = take((size_t)Bn * Tn * Hn);
  float* chb       = take((size_t)Bn * Tn * 2 * Hn);
  float* fc_out    = take((size_t)Bn * Tn * Hn);
  float* pg        = take((size_t)Bn * Tn * 3);
  (void)ws_size; (void)n_in; (void)in_sizes; (void)out_size;

  // ---- prologue ----
  k_embed<<<(Bn * Tn * En + 255) / 256, 256, 0, stream>>>(tokens, embed_w, xbuf);
  k_copy<<<(NLn * Bn * Hn + 255) / 256, 256, 0, stream>>>(hidden0, hid, NLn * Bn * Hn);

  // Precomputed projections (TDM + WMMA)
  gemm<0>(stream, cur_word_rep, attn_We, enc_proj, nullptr, Bn * Sn, Hn, Hn, Hn, Hn, Hn);
  gemm<0>(stream, ref_doc_reps, hd_We, doc_proj, nullptr, Bn * Rn, Hn, Hn, Hn, Hn, Hn);
  gemm<0>(stream, ref_word_reps, hw_We, word_proj, nullptr, Bn * Rn * Ln, Hn, Hn, Hn, Hn, Hn);

  // ---- sequential decode ----
  for (int t = 0; t < Tn; ++t) {
    const float* inp = xbuf + (size_t)t * En;   // x[:,t,:], row stride T*E
    int lda = Tn * En;
    for (int l = 0; l < NLn; ++l) {
      // gx = inp @ Wih^T + bih ; gh = h @ Whh^T + bhh   (WMMA, B transposed)
      gemm<1>(stream, inp, gru_wih + (size_t)l * H3 * Hn, gxb,
              gru_bih + (size_t)l * H3, Bn, H3, Hn, lda, Hn, H3);
      gemm<1>(stream, hid + (size_t)l * Bn * Hn, gru_whh + (size_t)l * H3 * Hn, ghb,
              gru_bhh + (size_t)l * H3, Bn, H3, Hn, Hn, Hn, H3);
      k_gru_gate<<<(Bn * Hn + 255) / 256, 256, 0, stream>>>(gxb, ghb, hid + (size_t)l * Bn * Hn);
      inp = hid + (size_t)l * Bn * Hn;
      lda = Hn;
    }
    const float* htop = hid + (size_t)2 * Bn * Hn;
    // decoder-state projections for the three attentions (WMMA)
    gemm<0>(stream, htop, attn_Wd, hpa, nullptr, Bn, Hn, Hn, Hn, Hn, Hn);
    gemm<0>(stream, htop, hd_Wd, hpd, nullptr, Bn, Hn, Hn, Hn, Hn, Hn);
    gemm<0>(stream, htop, hw_Wd, hpw, nullptr, Bn, Hn, Hn, Hn, Hn, Hn);
    // source attention + context (writes raw aw into output attn region)
    k_src_attn<<<Bn, 256, 0, stream>>>(enc_proj, hpa, attn_v, cur_word_rep, src_mask,
                                       ctx, out_attn, t);
    // hierarchical attention
    k_doc_attn<<<1, 128, 0, stream>>>(doc_proj, hpd, hd_v, ref_doc_mask, adw);
    k_word_score<<<(Bn * Rn * Ln + 255) / 256, 256, 0, stream>>>(word_proj, hpw, hw_v,
                                                                 ref_word_mask, scw);
    k_word_softmax_ra<<<Bn * Rn, Ln, 0, stream>>>(scw, adw, ra_all, t);
    k_rctx<<<(Bn * Hn + 255) / 256, 256, 0, stream>>>(ra_all, ref_word_reps, rctx, t);
    // fusion gate + combine
    k_gate_dot<<<1, 32, 0, stream>>>(ctx, rctx, gate_w, gate_b, gatev);
    k_combine<<<(Bn * Hn + 255) / 256, 256, 0, stream>>>(ctx, rctx, gatev, htop,
                                                         contexts, hiddens, t);
  }

  // ---- output head ----
  k_build_ch<<<(Bn * Tn * Hn + 255) / 256, 256, 0, stream>>>(contexts, hiddens, chb);
  gemm<0>(stream, chb, fc_w, fc_out, fc_b, Bn * Tn, Hn, 2 * Hn, 2 * Hn, Hn, Hn);
  // logits straight into d_out rows (ldc = V+OOV) -- dominant WMMA GEMM
  gemm<0>(stream, fc_out, out_w, out_final, out_b, Bn * Tn, Vn, Hn, Hn, Vn, Vn + OOVn);
  k_pgen<<<(Bn * Tn + 255) / 256, 256, 0, stream>>>(contexts, hiddens, xbuf, pgen_w, pgen_b, pg);
  k_vocab_softmax<<<Bn * Tn, 256, 0, stream>>>(out_final, pg);
  k_scatter_src<<<(Bn * Tn * Sn + 255) / 256, 256, 0, stream>>>(out_final, out_attn, pg, src_oov);
  k_scatter_ref<<<(Bn * Tn * Rn * Ln + 255) / 256, 256, 0, stream>>>(out_final, ra_all, pg, ref_oovs);
}
